// Informer_30958124270057
// MI455X (gfx1250) — matrix operand order repaired and checked by reading, not compiled, not run on hardware
//
#include <hip/hip_runtime.h>
#include <math.h>

#define BATCH     16
#define SEQ_LEN   512
#define LABEL_LEN 256
#define PRED_LEN  256
#define DMODEL    512
#define NHEADS    8
#define HEADE     64
#define DFF       2048

typedef __attribute__((ext_vector_type(16))) _Float16 v16h;
typedef __attribute__((ext_vector_type(8)))  _Float16 v8h;
typedef __attribute__((ext_vector_type(8)))  float    v8f;

static inline int ceil_div(int a, int b) { return (a + b - 1) / b; }

// ---------------------------------------------------------------------------
// Batched WMMA GEMM: C[z] = [act]( A[z](MxK) @ B(KxN) + bias ) (+C old)(+resid)
// REQUIRES: M % 128 == 0, N % 64 == 0, K % 32 == 0 (all call sites comply).
// Block = 256 threads = 8 waves; block tile 128(M) x 64(N); each wave owns a
// 32x32 tile -> 4 v_wmma per K-step with 2x2 fragment reuse. LDS is double
// buffered: one barrier per K-step, next tile's global loads overlap compute.
// A staged row-major [m][k]; B staged transposed [n][k] so both fragments are
// contiguous 8-half (ds_load_b128) chunks per the 16-bit WMMA VGPR layouts.
// ---------------------------------------------------------------------------
__global__ __launch_bounds__(256)
void k_gemm_wmma(const float* __restrict__ A, const float* __restrict__ Bm,
                 const float* __restrict__ bias, const float* __restrict__ resid,
                 float* __restrict__ C,
                 int M, int N, int K,
                 long long strideA, long long strideC, long long strideR,
                 int act, int accum)
{
    __shared__ _Float16 As[2][128][40];  // [buf][m][k], 80B rows (16B aligned)
    __shared__ _Float16 Bst[2][64][40];  // [buf][n][k] (transposed at store)
    const int bz = blockIdx.z;
    const float* Ab = A + (long long)bz * strideA;
    float* Cb = C + (long long)bz * strideC;
    const float* Rb = resid ? (resid + (long long)bz * strideR) : nullptr;
    const int m0 = blockIdx.y * 128;
    const int n0 = blockIdx.x * 64;
    const int tid  = threadIdx.x;
    const int wave = tid >> 5;
    const int lane = tid & 31;
    const int wm = (wave >> 1) * 32;   // 0,32,64,96
    const int wn = (wave & 1) * 32;    // 0,32
    const int hi = lane >> 4;
    const int lr = lane & 15;

    const int rowA = tid >> 1;         // 0..127
    const int segA = (tid & 1) * 16;   // 0 or 16
    const int kB   = tid >> 3;         // 0..31
    const int segB = (tid & 7) * 8;    // 0..56

    const float* apBase = Ab + (long long)(m0 + rowA) * K + segA;
    const float* bpBase = Bm + (long long)kB * N + (n0 + segB);

    float4 ar0, ar1, ar2, ar3, br0, br1;
    auto load_tile = [&](int k0) {
        const float* ap = apBase + k0;
        ar0 = *(const float4*)(ap + 0);
        ar1 = *(const float4*)(ap + 4);
        ar2 = *(const float4*)(ap + 8);
        ar3 = *(const float4*)(ap + 12);
        const float* bp = bpBase + (long long)k0 * N;
        br0 = *(const float4*)(bp + 0);
        br1 = *(const float4*)(bp + 4);
        if (k0 + 32 < K) { // prefetch following tile (global_prefetch_b8)
            __builtin_prefetch(ap + 32, 0, 1);
            __builtin_prefetch(bp + (long long)32 * N, 0, 1);
        }
    };
    auto store_tile = [&](int buf) {
        v8h v0, v1;
        v0[0] = (_Float16)ar0.x; v0[1] = (_Float16)ar0.y;
        v0[2] = (_Float16)ar0.z; v0[3] = (_Float16)ar0.w;
        v0[4] = (_Float16)ar1.x; v0[5] = (_Float16)ar1.y;
        v0[6] = (_Float16)ar1.z; v0[7] = (_Float16)ar1.w;
        v1[0] = (_Float16)ar2.x; v1[1] = (_Float16)ar2.y;
        v1[2] = (_Float16)ar2.z; v1[3] = (_Float16)ar2.w;
        v1[4] = (_Float16)ar3.x; v1[5] = (_Float16)ar3.y;
        v1[6] = (_Float16)ar3.z; v1[7] = (_Float16)ar3.w;
        *(v8h*)&As[buf][rowA][segA]     = v0;
        *(v8h*)&As[buf][rowA][segA + 8] = v1;
        Bst[buf][segB + 0][kB] = (_Float16)br0.x;
        Bst[buf][segB + 1][kB] = (_Float16)br0.y;
        Bst[buf][segB + 2][kB] = (_Float16)br0.z;
        Bst[buf][segB + 3][kB] = (_Float16)br0.w;
        Bst[buf][segB + 4][kB] = (_Float16)br1.x;
        Bst[buf][segB + 5][kB] = (_Float16)br1.y;
        Bst[buf][segB + 6][kB] = (_Float16)br1.z;
        Bst[buf][segB + 7][kB] = (_Float16)br1.w;
    };

    v8f acc[2][2] = {};
    load_tile(0);
    store_tile(0);
    const int nsteps = K >> 5;
    for (int s = 0; s < nsteps; ++s) {
        const int cur = s & 1;
        if (s + 1 < nsteps) load_tile((s + 1) * 32); // overlap with compute
        __syncthreads();                             // LDS[cur] ready
        v16h afr[2], bfr[2];
#pragma unroll
        for (int mi = 0; mi < 2; ++mi) {
            v8h alo = *(const v8h*)&As[cur][wm + 16 * mi + lr][hi * 8];
            v8h ahi = *(const v8h*)&As[cur][wm + 16 * mi + lr][16 + hi * 8];
#pragma unroll
            for (int j = 0; j < 8; ++j) { afr[mi][j] = alo[j]; afr[mi][j + 8] = ahi[j]; }
        }
#pragma unroll
        for (int ni = 0; ni < 2; ++ni) {
            v8h blo = *(const v8h*)&Bst[cur][wn + 16 * ni + lr][hi * 8];
            v8h bhi = *(const v8h*)&Bst[cur][wn + 16 * ni + lr][16 + hi * 8];
#pragma unroll
            for (int j = 0; j < 8; ++j) { bfr[ni][j] = blo[j]; bfr[ni][j + 8] = bhi[j]; }
        }
#pragma unroll
        for (int mi = 0; mi < 2; ++mi)
#pragma unroll
            for (int ni = 0; ni < 2; ++ni)
                acc[mi][ni] = __builtin_amdgcn_wmma_f32_16x16x32_f16(
                    false, afr[mi], false, bfr[ni], (short)0, acc[mi][ni], false, false);
        if (s + 1 < nsteps) store_tile(1 - cur);     // safe: last read pre-barrier
    }
#pragma unroll
    for (int mi = 0; mi < 2; ++mi)
#pragma unroll
        for (int ni = 0; ni < 2; ++ni)
#pragma unroll
            for (int rr = 0; rr < 8; ++rr) {
                int m = m0 + wm + 16 * mi + rr + 8 * hi;
                int n = n0 + wn + 16 * ni + lr;
                float v = acc[mi][ni][rr];
                if (bias)  v += bias[n];
                if (accum) v += Cb[(long long)m * N + n];
                if (act == 1) v = 0.5f * v * (1.0f + erff(v * 0.70710678118654752f));
                if (Rb)    v += Rb[(long long)m * N + n];
                Cb[(long long)m * N + n] = v;
            }
}

// ---------------------------------------------------------------------------
// LayerNorm over last dim D (one block per row)
// ---------------------------------------------------------------------------
__global__ __launch_bounds__(128)
void k_layernorm(const float* __restrict__ X, const float* __restrict__ g,
                 const float* __restrict__ b, float* __restrict__ Y, int D)
{
    const int row = blockIdx.x;
    const float* x = X + (long long)row * D;
    float* y = Y + (long long)row * D;
    __shared__ float red[128];
    float s = 0.0f;
    for (int i = threadIdx.x; i < D; i += 128) s += x[i];
    red[threadIdx.x] = s; __syncthreads();
    for (int t = 64; t > 0; t >>= 1) {
        if (threadIdx.x < t) red[threadIdx.x] += red[threadIdx.x + t];
        __syncthreads();
    }
    float mean = red[0] / (float)D;
    __syncthreads();
    float v = 0.0f;
    for (int i = threadIdx.x; i < D; i += 128) { float d = x[i] - mean; v += d * d; }
    red[threadIdx.x] = v; __syncthreads();
    for (int t = 64; t > 0; t >>= 1) {
        if (threadIdx.x < t) red[threadIdx.x] += red[threadIdx.x + t];
        __syncthreads();
    }
    float inv = rsqrtf(red[0] / (float)D + 1e-5f);
    for (int i = threadIdx.x; i < D; i += 128)
        y[i] = (x[i] - mean) * inv * g[i] + b[i];
}

// ---------------------------------------------------------------------------
// Token embedding: circular conv1d (in_ch=1, k=3) + sinusoidal positional enc
// ---------------------------------------------------------------------------
__global__ void k_emb_pos(const float* __restrict__ x, const float* __restrict__ w,
                          float* __restrict__ y, int L)
{
    long long gid = (long long)blockIdx.x * blockDim.x + threadIdx.x;
    long long total = (long long)BATCH * L * DMODEL;
    if (gid >= total) return;
    int d = (int)(gid % DMODEL);
    int l = (int)((gid / DMODEL) % L);
    int b = (int)(gid / ((long long)DMODEL * L));
    float acc = 0.0f;
    for (int t = 0; t < 3; ++t) {
        int ls = (l - 1 + t + L) % L;
        acc += w[d * 3 + t] * x[(long long)b * L + ls];
    }
    int i2 = d & ~1;
    float div = __expf((float)i2 * (-9.210340371976184f / (float)DMODEL));
    float ang = (float)l * div;
    acc += (d & 1) ? __cosf(ang) : __sinf(ang);
    y[gid] = acc;
}

__global__ void k_build_xdec(const float* __restrict__ xenc, float* __restrict__ xdec)
{
    int gid = blockIdx.x * blockDim.x + threadIdx.x;
    int total = BATCH * (LABEL_LEN + PRED_LEN);
    if (gid >= total) return;
    int l = gid % (LABEL_LEN + PRED_LEN);
    int b = gid / (LABEL_LEN + PRED_LEN);
    xdec[gid] = (l < LABEL_LEN) ? xenc[b * SEQ_LEN + (SEQ_LEN - LABEL_LEN) + l] : 0.0f;
}

// wrap-pad (B,L,D) -> (B,L+2,D)
__global__ void k_pad_wrap(const float* __restrict__ x, float* __restrict__ xp, int L)
{
    long long gid = (long long)blockIdx.x * blockDim.x + threadIdx.x;
    long long total = (long long)BATCH * (L + 2) * DMODEL;
    if (gid >= total) return;
    int d = (int)(gid % DMODEL);
    int l = (int)((gid / DMODEL) % (L + 2));
    int b = (int)(gid / ((long long)DMODEL * (L + 2)));
    int ls = (l - 1 + L) % L;
    xp[gid] = x[((long long)b * L + ls) * DMODEL + d];
}

// repack conv weight (dout,din,3) -> wt[t][din][dout]
__global__ void k_pack_w(const float* __restrict__ w, float* __restrict__ wt)
{
    int gid = blockIdx.x * blockDim.x + threadIdx.x;
    int total = 3 * DMODEL * DMODEL;
    if (gid >= total) return;
    int dout = gid % DMODEL;
    int din  = (gid / DMODEL) % DMODEL;
    int t    = gid / (DMODEL * DMODEL);
    wt[gid] = w[(dout * DMODEL + din) * 3 + t];
}

__global__ void k_bn_elu(float* __restrict__ y, const float* __restrict__ g,
                         const float* __restrict__ b, const float* __restrict__ rm,
                         const float* __restrict__ rv, long long total)
{
    long long gid = (long long)blockIdx.x * blockDim.x + threadIdx.x;
    if (gid >= total) return;
    int d = (int)(gid % DMODEL);
    float v = (y[gid] - rm[d]) * rsqrtf(rv[d] + 1e-5f) * g[d] + b[d];
    y[gid] = v > 0.0f ? v : (__expf(v) - 1.0f);
}

// maxpool1d window=3 stride=2 pad=1 over sequence dim
__global__ void k_maxpool(const float* __restrict__ y, float* __restrict__ o, int L)
{
    int Lo = L / 2;
    long long gid = (long long)blockIdx.x * blockDim.x + threadIdx.x;
    long long total = (long long)BATCH * Lo * DMODEL;
    if (gid >= total) return;
    int d  = (int)(gid % DMODEL);
    int lo = (int)((gid / DMODEL) % Lo);
    int b  = (int)(gid / ((long long)DMODEL * Lo));
    float m = -3.0e38f;
    for (int t = 0; t < 3; ++t) {
        int l = 2 * lo - 1 + t;
        if (l >= 0 && l < L) m = fmaxf(m, y[((long long)b * L + l) * DMODEL + d]);
    }
    o[gid] = m;
}

// ---------------------------------------------------------------------------
// ProbSparse attention helpers
// ---------------------------------------------------------------------------
__device__ inline unsigned hashu(unsigned x)
{
    x ^= x >> 16; x *= 0x7feb352dU; x ^= x >> 15; x *= 0x846ca68bU; x ^= x >> 16;
    return x;
}

__global__ void k_sample_M(const float* __restrict__ Q, const float* __restrict__ K,
                           float* __restrict__ Mout, int LQ, int LK, int Up, unsigned seed)
{
    int gid = blockIdx.x * blockDim.x + threadIdx.x;
    int total = BATCH * NHEADS * LQ;
    if (gid >= total) return;
    int l = gid % LQ;
    int h = (gid / LQ) % NHEADS;
    int b = gid / (LQ * NHEADS);
    const float* q = Q + ((long long)(b * LQ + l) * DMODEL) + h * HEADE;
    float mx = -3.0e38f, sm = 0.0f;
    for (int s = 0; s < Up; ++s) {
        int j = (int)(hashu(seed ^ (unsigned)(l * 131071 + s * 7919 + 17)) % (unsigned)LK);
        const float* kr = K + ((long long)(b * LK + j) * DMODEL) + h * HEADE;
        float dot = 0.0f;
        for (int e = 0; e < HEADE; ++e) dot += q[e] * kr[e];
        mx = fmaxf(mx, dot);
        sm += dot;
    }
    Mout[gid] = mx - sm / (float)LK;
}

__global__ void k_topk(const float* __restrict__ M, int* __restrict__ mtop, int LQ, int u)
{
    int bh = blockIdx.x;
    __shared__ float s[512];
    const float* row = M + (long long)bh * LQ;
    for (int i = threadIdx.x; i < LQ; i += blockDim.x) s[i] = row[i];
    __syncthreads();
    if (threadIdx.x == 0) {
        for (int t = 0; t < u; ++t) {
            float best = -3.0e38f; int bi = 0;
            for (int i = 0; i < LQ; ++i)
                if (s[i] > best) { best = s[i]; bi = i; }
            mtop[bh * u + t] = bi;
            s[bi] = -3.0e38f;
        }
    }
}

__global__ void k_vmean(const float* __restrict__ V, float* __restrict__ vm, int LK)
{
    int bh = blockIdx.x;
    int e = threadIdx.x;
    int b = bh / NHEADS, h = bh % NHEADS;
    float s = 0.0f;
    for (int j = 0; j < LK; ++j)
        s += V[((long long)(b * LK + j) * DMODEL) + h * HEADE + e];
    vm[bh * HEADE + e] = s / (float)LK;
}

__global__ void k_ctx_fill(const float* __restrict__ vm, float* __restrict__ ctx, int LQ)
{
    long long gid = (long long)blockIdx.x * blockDim.x + threadIdx.x;
    long long total = (long long)BATCH * LQ * DMODEL;
    if (gid >= total) return;
    int d = (int)(gid % DMODEL);
    int b = (int)(gid / ((long long)DMODEL * LQ));
    ctx[gid] = vm[(b * NHEADS + (d / HEADE)) * HEADE + (d & (HEADE - 1))];
}

__global__ void k_ctx_cumsum(const float* __restrict__ V, float* __restrict__ ctx, int L)
{
    int gid = blockIdx.x * blockDim.x + threadIdx.x;
    int total = BATCH * NHEADS * HEADE;
    if (gid >= total) return;
    int e = gid % HEADE;
    int h = (gid / HEADE) % NHEADS;
    int b = gid / (HEADE * NHEADS);
    float acc = 0.0f;
    for (int l = 0; l < L; ++l) {
        long long o = ((long long)(b * L + l) * DMODEL) + h * HEADE + e;
        acc += V[o];
        ctx[o] = acc;
    }
}

__global__ __launch_bounds__(128)
void k_attn_upd(const float* __restrict__ Q, const float* __restrict__ K,
                const float* __restrict__ V, const int* __restrict__ mtop,
                float* __restrict__ ctx, int LQ, int LK, int u, int maskflag)
{
    int g  = blockIdx.x;
    int ui = g % u;
    int bh = g / u;
    int h = bh % NHEADS, b = bh / NHEADS;
    int m = mtop[bh * u + ui];
    __shared__ float sc[512];
    __shared__ float red[128];
    const float* q = Q + ((long long)(b * LQ + m) * DMODEL) + h * HEADE;
    float lmax = -3.0e38f;
    for (int j = threadIdx.x; j < LK; j += 128) {
        const float* kr = K + ((long long)(b * LK + j) * DMODEL) + h * HEADE;
        float dot = 0.0f;
        for (int e = 0; e < HEADE; ++e) dot += q[e] * kr[e];
        dot *= 0.125f; // 1/sqrt(64)
        if (maskflag && j > m) dot = -1.0e30f;
        sc[j] = dot;
        lmax = fmaxf(lmax, dot);
    }
    red[threadIdx.x] = lmax; __syncthreads();
    for (int t = 64; t > 0; t >>= 1) {
        if (threadIdx.x < t) red[threadIdx.x] = fmaxf(red[threadIdx.x], red[threadIdx.x + t]);
        __syncthreads();
    }
    float mx = red[0];
    __syncthreads();
    float lsum = 0.0f;
    for (int j = threadIdx.x; j < LK; j += 128) {
        float e = __expf(sc[j] - mx);
        sc[j] = e;
        lsum += e;
    }
    red[threadIdx.x] = lsum; __syncthreads();
    for (int t = 64; t > 0; t >>= 1) {
        if (threadIdx.x < t) red[threadIdx.x] += red[threadIdx.x + t];
        __syncthreads();
    }
    float inv = 1.0f / red[0];
    if (threadIdx.x < HEADE) {
        int e = threadIdx.x;
        float acc = 0.0f;
        for (int j = 0; j < LK; ++j)
            acc += sc[j] * V[((long long)(b * LK + j) * DMODEL) + h * HEADE + e];
        ctx[((long long)(b * LQ + m) * DMODEL) + h * HEADE + e] = acc * inv;
    }
}

__global__ void k_proj(const float* __restrict__ X, const float* __restrict__ w,
                       const float* __restrict__ b0, float* __restrict__ out)
{
    int gid = blockIdx.x * blockDim.x + threadIdx.x;
    int total = BATCH * PRED_LEN;
    if (gid >= total) return;
    int lp = gid % PRED_LEN;
    int b = gid / PRED_LEN;
    const float* x = X + (long long)(b * (LABEL_LEN + PRED_LEN) + LABEL_LEN + lp) * DMODEL;
    float s = b0[0];
    for (int i = 0; i < DMODEL; ++i) s += x[i] * w[i];
    out[gid] = s;
}

// ---------------------------------------------------------------------------
// Host-side orchestration
// ---------------------------------------------------------------------------
static void gemm(hipStream_t st, const float* A, const float* B, const float* bias,
                 const float* resid, float* C, int M, int N, int K,
                 int batch, long long sA, long long sC, long long sR,
                 int act, int accum)
{
    dim3 g(ceil_div(N, 64), ceil_div(M, 128), batch);
    k_gemm_wmma<<<g, 256, 0, st>>>(A, B, bias, resid, C, M, N, K, sA, sC, sR, act, accum);
}

struct AttnW { const float *Wk, *Wo, *Wq, *Wv, *bk, *bo, *bq, *bv; };

static int factor_u(int L)
{
    int c = (int)ceilf(logf((float)L));
    int v = 5 * c;
    return v < L ? v : L;
}

struct Scratch {
    float *Q, *K, *V, *Ctx, *Mbuf, *vmean;
    int *mtop;
};

static void attention(hipStream_t st, const AttnW& w, const float* Xq, const float* Xkv,
                      float* Xout, int LQ, int LK, int maskflag, unsigned seed,
                      const Scratch& s)
{
    int u  = factor_u(LQ);
    int Up = factor_u(LK);
    int Mq = BATCH * LQ, Mk = BATCH * LK;
    gemm(st, Xq,  w.Wq, w.bq, nullptr, s.Q, Mq, DMODEL, DMODEL, 1, 0, 0, 0, 0, 0);
    gemm(st, Xkv, w.Wk, w.bk, nullptr, s.K, Mk, DMODEL, DMODEL, 1, 0, 0, 0, 0, 0);
    gemm(st, Xkv, w.Wv, w.bv, nullptr, s.V, Mk, DMODEL, DMODEL, 1, 0, 0, 0, 0, 0);
    int totM = BATCH * NHEADS * LQ;
    k_sample_M<<<ceil_div(totM, 256), 256, 0, st>>>(s.Q, s.K, s.Mbuf, LQ, LK, Up, seed);
    k_topk<<<BATCH * NHEADS, 64, 0, st>>>(s.Mbuf, s.mtop, LQ, u);
    if (maskflag) {
        k_ctx_cumsum<<<ceil_div(BATCH * NHEADS * HEADE, 256), 256, 0, st>>>(s.V, s.Ctx, LQ);
    } else {
        k_vmean<<<BATCH * NHEADS, HEADE, 0, st>>>(s.V, s.vmean, LK);
        int tot = BATCH * LQ * DMODEL;
        k_ctx_fill<<<ceil_div(tot, 256), 256, 0, st>>>(s.vmean, s.Ctx, LQ);
    }
    k_attn_upd<<<BATCH * NHEADS * u, 128, 0, st>>>(s.Q, s.K, s.V, s.mtop, s.Ctx, LQ, LK, u, maskflag);
    gemm(st, s.Ctx, w.Wo, w.bo, Xq, Xout, Mq, DMODEL, DMODEL, 1, 0, 0, 0, 0, 0);
}

static void ffn(hipStream_t st, const float* c1w, const float* c1b,
                const float* c2w, const float* c2b,
                const float* X, float* bufF, float* Y, int Mrows)
{
    gemm(st, X, c1w, c1b, nullptr, bufF, Mrows, DFF, DMODEL, 1, 0, 0, 0, 1 /*gelu*/, 0);
    gemm(st, bufF, c2w, c2b, X, Y, Mrows, DMODEL, DFF, 1, 0, 0, 0, 0, 0);
}

extern "C" void kernel_launch(void* const* d_in, const int* in_sizes, int n_in,
                              void* d_out, int out_size, void* d_ws, size_t ws_size,
                              hipStream_t stream)
{
    (void)in_sizes; (void)n_in; (void)out_size; (void)ws_size;
    // pytree (sorted dict key) flattening order of setup_inputs() — see prior rounds.
    const float* in[95];
    for (int i = 0; i < 95; ++i) in[i] = (const float*)d_in[i];
    const float* x_enc = in[94];

    // ---- workspace carve (floats) ----
    float* p = (float*)d_ws;
    auto alloc = [&](long long n) { float* r = p; p += n; return r; };
    const long long ACT = (long long)BATCH * SEQ_LEN * DMODEL; // 4 Mi floats
    float* bufX = alloc(ACT);
    float* bufY = alloc(ACT);
    float* bufD = alloc(ACT);
    Scratch s;
    s.Q   = alloc(ACT);
    s.K   = alloc(ACT);
    s.V   = alloc(ACT);
    s.Ctx = alloc(ACT);
    float* bufF = alloc((long long)BATCH * SEQ_LEN * DFF);   // 16 Mi floats
    float* xp = bufF;                                        // alias (conv vs ffn exclusive)
    float* wt = bufF + (long long)BATCH * (SEQ_LEN + 2) * DMODEL;
    s.Mbuf  = alloc((long long)BATCH * NHEADS * SEQ_LEN);
    s.mtop  = (int*)alloc(8192);
    s.vmean = alloc((long long)BATCH * NHEADS * HEADE);
    float* xdec = alloc((long long)BATCH * (LABEL_LEN + PRED_LEN));

    // ---- encoder embedding ----
    {
        int tot = BATCH * SEQ_LEN * DMODEL;
        k_emb_pos<<<ceil_div(tot, 256), 256, 0, stream>>>(x_enc, in[41], bufX, SEQ_LEN);
    }

    // ---- encoder layers with distillation ----
    int L = SEQ_LEN;
    for (int li = 0; li < 3; ++li) {
        const float* const* P = in + 42 + 16 * li;
        AttnW w{P[0], P[1], P[2], P[3], P[4], P[5], P[6], P[7]};
        attention(stream, w, bufX, bufX, bufY, L, L, 0, 0xC0FFEEu + li, s);
        k_layernorm<<<BATCH * L, 128, 0, stream>>>(bufY, P[13], P[12], bufX, DMODEL);
        ffn(stream, P[9], P[8], P[11], P[10], bufX, bufF, bufY, BATCH * L);
        k_layernorm<<<BATCH * L, 128, 0, stream>>>(bufY, P[15], P[14], bufX, DMODEL);
        if (li < 2) {
            const float* const* CL = in + 6 * li; // b,bn_b,bn_g,bn_rm,bn_rv,w
            k_pack_w<<<ceil_div(3 * DMODEL * DMODEL, 256), 256, 0, stream>>>(CL[5], wt);
            int totp = BATCH * (L + 2) * DMODEL;
            k_pad_wrap<<<ceil_div(totp, 256), 256, 0, stream>>>(bufX, xp, L);
            for (int t = 0; t < 3; ++t)
                gemm(stream, xp + t * DMODEL, wt + (long long)t * DMODEL * DMODEL,
                     (t == 0 ? CL[0] : nullptr), nullptr, bufY, L, DMODEL, DMODEL,
                     BATCH, (long long)(L + 2) * DMODEL, (long long)L * DMODEL, 0,
                     0, (t > 0 ? 1 : 0));
            long long tot = (long long)BATCH * L * DMODEL;
            k_bn_elu<<<ceil_div((int)tot, 256), 256, 0, stream>>>(bufY, CL[2], CL[1], CL[3], CL[4], tot);
            int Lo = L / 2;
            int toto = BATCH * Lo * DMODEL;
            k_maxpool<<<ceil_div(toto, 256), 256, 0, stream>>>(bufY, bufX, L);
            L = Lo;
        }
    }
    // final encoder norm (in-place); bufX now holds enc output (B, 128, 512)
    k_layernorm<<<BATCH * L, 128, 0, stream>>>(bufX, in[91], in[90], bufX, DMODEL);
    const int LE = L; // 128

    // ---- decoder ----
    const int LD = LABEL_LEN + PRED_LEN; // 512
    {
        int tot = BATCH * LD;
        k_build_xdec<<<ceil_div(tot, 256), 256, 0, stream>>>(x_enc, xdec);
        int tote = BATCH * LD * DMODEL;
        k_emb_pos<<<ceil_div(tote, 256), 256, 0, stream>>>(xdec, in[12], bufD, LD);
    }
    AttnW wself{in[31], in[32], in[33], in[34], in[35], in[36], in[37], in[38]};
    attention(stream, wself, bufD, bufD, bufY, LD, LD, 1, 0xDEC0DEu, s);
    k_layernorm<<<BATCH * LD, 128, 0, stream>>>(bufY, in[26], in[25], bufD, DMODEL);
    AttnW wcross{in[17], in[18], in[19], in[20], in[21], in[22], in[23], in[24]};
    attention(stream, wcross, bufD, bufX, bufY, LD, LE, 0, 0xCAB1E5u, s);
    k_layernorm<<<BATCH * LD, 128, 0, stream>>>(bufY, in[28], in[27], bufD, DMODEL);
    ffn(stream, in[14], in[13], in[16], in[15], bufD, bufF, bufY, BATCH * LD);
    k_layernorm<<<BATCH * LD, 128, 0, stream>>>(bufY, in[30], in[29], bufD, DMODEL);
    k_layernorm<<<BATCH * LD, 128, 0, stream>>>(bufD, in[40], in[39], bufD, DMODEL);

    // ---- output projection ----
    k_proj<<<ceil_div(BATCH * PRED_LEN, 256), 256, 0, stream>>>(bufD, in[93], in[92], (float*)d_out);
}
